// MAB_84980222919283
// MI455X (gfx1250) — matrix operand order repaired and checked by reading, not compiled, not run on hardware
//
#include <hip/hip_runtime.h>
#include <hip/hip_bf16.h>

typedef __attribute__((ext_vector_type(8)))  _Float16 v8h;
typedef __attribute__((ext_vector_type(16))) _Float16 v16h;
typedef __attribute__((ext_vector_type(8)))  float    v8f;

#define BB 16
#define TT 12
#define NN 325
#define FIN 11
#define HH 8
#define DHH 64
#define DD 512
#define NPAD 352          // N padded to 11 K-tiles of 32
#define SCOLS 336         // 21 tiles of 16 score columns
#define STRIPS 21         // ceil(325/16) q-row strips
#define JT 21             // k column tiles of 16
#define KTILES 11         // P@V K-tiles of 32
#define SLICES (BB*TT*HH) // 1536
#define QKV_ELEMS ((size_t)SLICES * NPAD * DHH)

// ---------------------------------------------------------------------------
// WMMA fragment loaders.
// A-matrix 16x32 f16 (ISA 7.12.2): lanes 0-15 row M=l hold K0-7 (v0-3) and
// K16-23 (v4-7); lanes 16-31 hold K8-15 / K24-31.  => per lane: two 8-half
// contiguous chunks at +half*8 and +half*8+16.
__device__ __forceinline__ v16h load_afrag(const _Float16* base, int stride,
                                           int row, int k0, int half) {
  const _Float16* p = base + (size_t)row * stride + k0 + half * 8;
  v16h a;
  *(v8h*)&a       = *(const v8h*)(p);
  *((v8h*)&a + 1) = *(const v8h*)(p + 16);
  return a;
}
// B-matrix 32x16 f16: lane = column; same K order as A: lanes 0-15 hold
// K0-7 / K16-23, lanes 16-31 hold K8-15 / K24-31.  Loaded from B^T stored
// row-major [N][Kstride] => two 8-half chunks at +half*8 and +half*8+16.
__device__ __forceinline__ v16h load_bfrag(const _Float16* base, int stride,
                                           int col, int k0, int half) {
  const _Float16* p = base + (size_t)col * stride + k0 + half * 8;
  v16h b;
  *(v8h*)&b       = *(const v8h*)(p);
  *((v8h*)&b + 1) = *(const v8h*)(p + 16);
  return b;
}

// ---------------------------------------------------------------------------
// Kernel 1: q = relu(Q@Wq+bq), k = relu(K@Wk+bk), v = relu(K@Wv+bv).
// K=11 inner dim -> VALU.  Writes f16:
//   qh,kh : [slice][NPAD][64]  (row-major, rows >= 325 zeroed)
//   vT    : [slice][64][NPAD]  (transposed, cols >= 325 zeroed)
__global__ __launch_bounds__(256) void proj_kernel(
    const float* __restrict__ Q, const float* __restrict__ Kin,
    const float* __restrict__ Wq, const float* __restrict__ bq,
    const float* __restrict__ Wk, const float* __restrict__ bk,
    const float* __restrict__ Wv, const float* __restrict__ bv,
    _Float16* __restrict__ qh, _Float16* __restrict__ kh,
    _Float16* __restrict__ vT) {
  int blk = blockIdx.x;
  int n   = blk % NPAD;
  int bt  = blk / NPAD;          // b*T + t
  bool valid = (n < NN);
  float xq[FIN], xk[FIN];
  if (valid) {
    const float* qrow = Q   + ((size_t)bt * NN + n) * FIN;
    const float* krow = Kin + ((size_t)bt * NN + n) * FIN;
#pragma unroll
    for (int i = 0; i < FIN; ++i) { xq[i] = qrow[i]; xk[i] = krow[i]; }
  }
#pragma unroll
  for (int rep = 0; rep < 2; ++rep) {
    int d  = threadIdx.x + rep * 256;
    int h  = d >> 6, dh = d & 63;
    int s  = bt * HH + h;
    size_t qidx = ((size_t)s * NPAD + n) * DHH + dh;
    size_t vidx = ((size_t)s * DHH + dh) * NPAD + n;
    float aq = 0.f, ak = 0.f, av = 0.f;
    if (valid) {
      aq = bq[d]; ak = bk[d]; av = bv[d];
#pragma unroll
      for (int i = 0; i < FIN; ++i) {
        aq += xq[i] * Wq[i * DD + d];
        ak += xk[i] * Wk[i * DD + d];
        av += xk[i] * Wv[i * DD + d];
      }
      aq = fmaxf(aq, 0.f); ak = fmaxf(ak, 0.f); av = fmaxf(av, 0.f);
    }
    qh[qidx] = (_Float16)aq;
    kh[qidx] = (_Float16)ak;
    vT[vidx] = (_Float16)av;
  }
}

// ---------------------------------------------------------------------------
// Kernel 2: per (slice, 16-row q strip): S = qk^T/8, softmax, attn out,
// O = P@v.  One wave (32 threads) per block; all matmuls via
// v_wmma_f32_16x16x32_f16.
__global__ __launch_bounds__(32) void attn_kernel(
    const _Float16* __restrict__ qh, const _Float16* __restrict__ kh,
    const _Float16* __restrict__ vT,
    float* __restrict__ attn_out, float* __restrict__ Ows) {
  __shared__ __align__(16) float    S[16][SCOLS];   // scores / probs (f32)
  __shared__ __align__(16) _Float16 P[16][NPAD];    // probs as f16 A-matrix

  const int lane  = threadIdx.x;
  const int half  = lane >> 4;
  const int l     = lane & 15;
  const int sid   = blockIdx.x / STRIPS;
  const int strip = blockIdx.x % STRIPS;
  const int bt = sid / HH, h = sid % HH;
  const int b  = bt / TT,  t = bt % TT;

  const _Float16* qbase = qh + (size_t)sid * NPAD * DHH;
  const _Float16* kbase = kh + (size_t)sid * NPAD * DHH;
  const _Float16* vbase = vT + (size_t)sid * DHH * NPAD;

  // Q strip A-fragments (DH=64 -> two K-tiles of 32)
  const int qrow = strip * 16 + l;
  v16h a0 = load_afrag(qbase, DHH, qrow, 0,  half);
  v16h a1 = load_afrag(qbase, DHH, qrow, 32, half);

  // --- scores: 21 column tiles, 2 WMMA each, scaled into LDS ---
  for (int j = 0; j < JT; ++j) {
    v16h b0 = load_bfrag(kbase, DHH, j * 16 + l, 0,  half);
    v16h b1 = load_bfrag(kbase, DHH, j * 16 + l, 32, half);
    v8f c = {};
    c = __builtin_amdgcn_wmma_f32_16x16x32_f16(false, a0, false, b0,
                                               (short)0, c, false, false);
    c = __builtin_amdgcn_wmma_f32_16x16x32_f16(false, a1, false, b1,
                                               (short)0, c, false, false);
#pragma unroll
    for (int r = 0; r < 8; ++r)                       // C layout: lane=col,
      S[half * 8 + r][j * 16 + l] = c[r] * 0.125f;    // vgpr=row(+8 hi lanes)
  }

  // zero the f16 prob matrix (covers K-padding 325..351 and invalid rows)
  for (int idx = lane; idx < 16 * NPAD; idx += 32)
    ((_Float16*)P)[idx] = (_Float16)0.f;
  __syncthreads();

  // --- softmax: one lane per row over 325 valid columns ---
  if (lane < 16) {
    int gq = strip * 16 + lane;
    if (gq < NN) {
      float m = -3.4e38f;
      for (int c = 0; c < NN; ++c) m = fmaxf(m, S[lane][c]);
      float sum = 0.f;
      for (int c = 0; c < NN; ++c) {
        float e = __expf(S[lane][c] - m);
        S[lane][c] = e; sum += e;
      }
      float inv = 1.f / sum;
      for (int c = 0; c < NN; ++c) {
        float p = S[lane][c] * inv;
        S[lane][c] = p;
        P[lane][c] = (_Float16)p;
      }
    }
  }
  __syncthreads();

  // --- write attn output [H*B, T, N, N] ---
  for (int r = 0; r < 16; ++r) {
    int gq = strip * 16 + r;
    if (gq >= NN) break;
    float* dst = attn_out + ((((size_t)(h * BB + b) * TT + t) * NN + gq) * NN);
    for (int c = lane; c < NN; c += 32) dst[c] = S[r][c];
  }

  // --- O = P @ v : 11 K-tiles x 4 DH column tiles = 44 WMMAs ---
  v8f acc[4] = {};
  for (int kt = 0; kt < KTILES; ++kt) {
    v16h pa = load_afrag(&P[0][0], NPAD, l, kt * 32, half);
#pragma unroll
    for (int ct = 0; ct < 4; ++ct) {
      v16h vb = load_bfrag(vbase, NPAD, ct * 16 + l, kt * 32, half);
      acc[ct] = __builtin_amdgcn_wmma_f32_16x16x32_f16(false, pa, false, vb,
                                                       (short)0, acc[ct],
                                                       false, false);
    }
  }

  // store O as f32 [B*T][N][512] merged over heads for the final projection
#pragma unroll
  for (int ct = 0; ct < 4; ++ct) {
#pragma unroll
    for (int r = 0; r < 8; ++r) {
      int gq = strip * 16 + half * 8 + r;
      if (gq < NN)
        Ows[((size_t)bt * NN + gq) * DD + h * DHH + ct * 16 + l] = acc[ct][r];
    }
  }
}

// ---------------------------------------------------------------------------
// Kernel 3: out = relu(O @ Wo + bo), 512 -> 11.
__global__ __launch_bounds__(256) void outproj_kernel(
    const float* __restrict__ O, const float* __restrict__ Wo,
    const float* __restrict__ bo, float* __restrict__ out) {
  int tid = blockIdx.x * 256 + threadIdx.x;
  if (tid >= BB * TT * NN * FIN) return;
  int row = tid / FIN, oc = tid % FIN;
  const float* orow = O + (size_t)row * DD;
  float acc = bo[oc];
  for (int d = 0; d < DD; ++d) acc += orow[d] * Wo[d * FIN + oc];
  out[tid] = fmaxf(acc, 0.f);
}

// ---------------------------------------------------------------------------
extern "C" void kernel_launch(void* const* d_in, const int* in_sizes, int n_in,
                              void* d_out, int out_size, void* d_ws,
                              size_t ws_size, hipStream_t stream) {
  (void)in_sizes; (void)n_in; (void)out_size; (void)ws_size;
  const float* Q   = (const float*)d_in[0];
  const float* Kin = (const float*)d_in[1];
  // d_in[2] = batch_size scalar (compile-time constant here)
  const float* Wq  = (const float*)d_in[3];
  const float* bq  = (const float*)d_in[4];
  const float* Wk  = (const float*)d_in[5];
  const float* bk  = (const float*)d_in[6];
  const float* Wv  = (const float*)d_in[7];
  const float* bv  = (const float*)d_in[8];
  const float* Wo  = (const float*)d_in[9];
  const float* bo  = (const float*)d_in[10];

  _Float16* qh = (_Float16*)d_ws;
  _Float16* kh = qh + QKV_ELEMS;
  _Float16* vT = kh + QKV_ELEMS;
  float*    Ows = (float*)(vT + QKV_ELEMS);

  float* out  = (float*)d_out;
  float* attn = out + (size_t)BB * TT * NN * FIN;   // tuple: (out, attn)

  proj_kernel<<<BB * TT * NPAD, 256, 0, stream>>>(Q, Kin, Wq, bq, Wk, bk,
                                                  Wv, bv, qh, kh, vT);
  attn_kernel<<<SLICES * STRIPS, 32, 0, stream>>>(qh, kh, vT, attn, Ows);
  outproj_kernel<<<(BB * TT * NN * FIN + 255) / 256, 256, 0, stream>>>(
      Ows, Wo, bo, out);
}
